// Encoder_72541997629632
// MI455X (gfx1250) — matrix operand order repaired and checked by reading, not compiled, not run on hardware
//
#include <hip/hip_runtime.h>
#include <hip/hip_bf16.h>
#include <stdint.h>

// ---------------- types ----------------
typedef _Float16 v16h __attribute__((ext_vector_type(16)));
typedef _Float16 half8 __attribute__((ext_vector_type(8)));
typedef float    v8f  __attribute__((ext_vector_type(8)));

#define TT   2048   // sequence length
#define BB   64     // batch
#define HH   128    // hidden
#define DD   128    // embedding dim
#define G4   512    // 4*H

// workspace layout (bytes)
#define WS_WIH16   0                       // 512*128 f16 = 131072
#define WS_WHH16   131072                  // 512*128 f16 = 131072
#define WS_BIAS    262144                  // 512 f32     = 2048
#define WS_XG      264192                  // T*B*4H f16  = 134217728  (tile-interleaved layout)

// xg layout (in halves): [t][tile(0..31)][g(0..3)][lane(0..31)][v(0..7)]
__device__ __forceinline__ size_t xg_off_h(int t, int tile, int g, int lane) {
    return ((((size_t)t * 32 + tile) * 4 + g) * 256) + (size_t)lane * 8;
}

__device__ __forceinline__ v16h cat16(half8 lo, half8 hi) {
    return __builtin_shufflevector(lo, hi, 0,1,2,3,4,5,6,7,8,9,10,11,12,13,14,15);
}
__device__ __forceinline__ float sigf(float x) { return 1.0f / (1.0f + __expf(-x)); }

// ---------------- prep: f32 -> f16 weights, fused bias ----------------
__global__ void prep_kernel(const float* __restrict__ wih, const float* __restrict__ whh,
                            const float* __restrict__ bih, const float* __restrict__ bhh,
                            _Float16* __restrict__ wih16, _Float16* __restrict__ whh16,
                            float* __restrict__ bias) {
    int idx = blockIdx.x * blockDim.x + threadIdx.x;
    if (idx < 65536)            wih16[idx]         = (_Float16)wih[idx];
    else if (idx < 131072)      whh16[idx - 65536] = (_Float16)whh[idx - 65536];
    else if (idx < 131072+512)  { int i = idx - 131072; bias[i] = bih[i] + bhh[i]; }
}

// ---------------- phase 1: xg[t] = emb[tok]·W_ih^T + b  (one block per t) ----------------
__global__ void __launch_bounds__(1024, 1)
xgates_kernel(const int* __restrict__ tokens, const float* __restrict__ emb,
              const _Float16* __restrict__ wih16, const float* __restrict__ bias,
              _Float16* __restrict__ xg) {
    extern __shared__ char smem[];
    _Float16* x_lds   = (_Float16*)smem;            // 64*128 f16 = 16384 B
    float*    bias_ld = (float*)(smem + 16384);     // 512 f32   = 2048 B

    const int t    = blockIdx.x;
    const int tid  = threadIdx.x;
    const int wave = tid >> 5, lane = tid & 31;
    const int mt   = wave & 3,  j   = wave >> 2;      // row-block 0..3, col-block 0..7
    const int tile = mt * 8 + j;
    const int lh   = lane & 15, lg = lane >> 4;

    // stage embeddings for this timestep (gather + f32->f16)
    {
        int b  = tid >> 4;
        int k0 = (tid & 15) * 8;
        int tok = tokens[(size_t)b * TT + t];
        const float* er = emb + (size_t)tok * DD + k0;
        float4 f0 = *(const float4*)er;
        float4 f1 = *(const float4*)(er + 4);
        half8 h;
        h[0]=(_Float16)f0.x; h[1]=(_Float16)f0.y; h[2]=(_Float16)f0.z; h[3]=(_Float16)f0.w;
        h[4]=(_Float16)f1.x; h[5]=(_Float16)f1.y; h[6]=(_Float16)f1.z; h[7]=(_Float16)f1.w;
        *(half8*)(x_lds + b * DD + k0) = h;
    }
    if (tid < 512) bias_ld[tid] = bias[tid];
    __syncthreads();

    v8f acc[4];
    #pragma unroll
    for (int g = 0; g < 4; ++g) acc[g] = (v8f){0,0,0,0,0,0,0,0};

    #pragma unroll
    for (int kc = 0; kc < 4; ++kc) {
        // A tile: x rows 16*mt..+15, K chunk 32*kc..+31 (ISA 16-bit A layout)
        int ar = 16 * mt + lh;
        int ak = 32 * kc + lg * 8;
        half8 alo = *(const half8*)(x_lds + ar * DD + ak);
        half8 ahi = *(const half8*)(x_lds + ar * DD + ak + 16);
        v16h A = cat16(alo, ahi);
        #pragma unroll
        for (int g = 0; g < 4; ++g) {
            // B tile: W_ih rows (gate cols) 128g+16j+lh, K = 8*lg+{0..7} and 16+8*lg+{0..7} per lane
            const _Float16* bp = wih16 + (size_t)(128 * g + 16 * j + lh) * DD + 32 * kc + lg * 8;
            v16h Bv = cat16(*(const half8*)bp, *(const half8*)(bp + 16));
            acc[g] = __builtin_amdgcn_wmma_f32_16x16x32_f16(
                false, A, false, Bv, (short)0, acc[g], false, false);
        }
    }

    #pragma unroll
    for (int g = 0; g < 4; ++g) {
        float bv = bias_ld[128 * g + 16 * j + lh];
        half8 o;
        #pragma unroll
        for (int v = 0; v < 8; ++v) o[v] = (_Float16)(acc[g][v] + bv);
        *(half8*)(xg + xg_off_h(t, tile, g, lane)) = o;
    }
}

// ---------------- phase 2: persistent single-WGP LSTM scan ----------------
// LDS: W_hh f16 (131072) + h f16 (16384) + xg double buffer (2*65536) = 278528 B  (<= 320 KB WGP LDS)
#define L_WHH  0
#define L_H    131072
#define L_XG   147456
#define SMEM2  278528

__global__ void __launch_bounds__(1024, 1)
scan_kernel(const _Float16* __restrict__ whh16, const _Float16* __restrict__ xg,
            float* __restrict__ out) {
    extern __shared__ char smem[];
    _Float16* whh_l = (_Float16*)(smem + L_WHH);
    _Float16* h_l   = (_Float16*)(smem + L_H);
    _Float16* xg_l  = (_Float16*)(smem + L_XG);

    const int tid  = threadIdx.x;
    const int wave = tid >> 5, lane = tid & 31;
    const int mt   = wave & 3,  j   = wave >> 2;
    const int tile = mt * 8 + j;
    const int lh   = lane & 15, lg = lane >> 4;

    // copy W_hh (f16) global -> LDS; zero h
    #pragma unroll
    for (int it = 0; it < 8; ++it) {
        int i = tid * 8 + it * 8192;                 // halves
        *(int4*)(smem + L_WHH + i * 2) = *(const int4*)((const char*)whh16 + i * 2);
    }
    *(int4*)(smem + L_H + tid * 16) = (int4){0,0,0,0};

    float c[8], hreg[8];
    #pragma unroll
    for (int v = 0; v < 8; ++v) { c[v] = 0.f; hreg[v] = 0.f; }

    // kick off async LDS fill of xg[t=0] into buffer 0 (ASYNCcnt-tracked)
    #pragma unroll
    for (int g = 0; g < 4; ++g) {
        uint32_t dst = (uint32_t)(uintptr_t)(smem + L_XG + (tile * 4 + g) * 512 + lane * 16);
        uint64_t src = (uint64_t)(uintptr_t)(xg + xg_off_h(0, tile, g, lane));
        asm volatile("global_load_async_to_lds_b128 %0, %1, off" :: "v"(dst), "v"(src) : "memory");
    }
    __syncthreads();

    for (int t = 0; t < TT; ++t) {
        const int cb = t & 1, nb = cb ^ 1;
        const int tn = (t + 1) & (TT - 1);

        // wait for xg[t] (issued last iteration), then prefetch xg[t+1] asynchronously
        asm volatile("s_wait_asynccnt 0" ::: "memory");
        #pragma unroll
        for (int g = 0; g < 4; ++g) {
            uint32_t dst = (uint32_t)(uintptr_t)(smem + L_XG + nb * 65536 + (tile * 4 + g) * 512 + lane * 16);
            uint64_t src = (uint64_t)(uintptr_t)(xg + xg_off_h(tn, tile, g, lane));
            asm volatile("global_load_async_to_lds_b128 %0, %1, off" :: "v"(dst), "v"(src) : "memory");
        }

        // recurrent GEMM: gates += h(t) · W_hh^T, four gate tiles per wave
        v8f acc[4];
        #pragma unroll
        for (int g = 0; g < 4; ++g) acc[g] = (v8f){0,0,0,0,0,0,0,0};
        #pragma unroll
        for (int kc = 0; kc < 4; ++kc) {
            int ar = 16 * mt + lh;
            int ak = 32 * kc + lg * 8;
            half8 alo = *(const half8*)(h_l + ar * HH + ak);
            half8 ahi = *(const half8*)(h_l + ar * HH + ak + 16);
            v16h A = cat16(alo, ahi);
            #pragma unroll
            for (int g = 0; g < 4; ++g) {
                const _Float16* bp = whh_l + (128 * g + 16 * j + lh) * HH + 32 * kc + lg * 8;
                v16h Bv = cat16(*(const half8*)bp, *(const half8*)(bp + 16));
                acc[g] = __builtin_amdgcn_wmma_f32_16x16x32_f16(
                    false, A, false, Bv, (short)0, acc[g], false, false);
            }
        }
        __syncthreads();   // all waves done reading h(t)

        // LSTM elementwise: this wave owns all 4 gates of its 16x16 (b,n) tile
        half8 xcur[4];
        #pragma unroll
        for (int g = 0; g < 4; ++g)
            xcur[g] = *(const half8*)(xg_l + cb * 32768 + (tile * 4 + g) * 256 + lane * 8);

        #pragma unroll
        for (int v = 0; v < 8; ++v) {
            float gi = acc[0][v] + (float)xcur[0][v];
            float gf = acc[1][v] + (float)xcur[1][v];
            float gg = acc[2][v] + (float)xcur[2][v];
            float go = acc[3][v] + (float)xcur[3][v];
            float cn = sigf(gf) * c[v] + sigf(gi) * tanhf(gg);
            c[v] = cn;
            hreg[v] = sigf(go) * tanhf(cn);
            int b = 16 * mt + lg * 8 + v;   // D layout: M = 8*(lane/16)+v
            int n = 16 * j + lh;            //           N = lane%16
            h_l[b * HH + n] = (_Float16)hreg[v];
        }
        __syncthreads();   // h(t+1) visible before next GEMM
    }

    // out = [h (B*H), c (B*H)] f32
    #pragma unroll
    for (int v = 0; v < 8; ++v) {
        int b = 16 * mt + lg * 8 + v;
        int n = 16 * j + lh;
        out[b * HH + n]        = hreg[v];
        out[8192 + b * HH + n] = c[v];
    }
}

// ---------------- launcher ----------------
extern "C" void kernel_launch(void* const* d_in, const int* in_sizes, int n_in,
                              void* d_out, int out_size, void* d_ws, size_t ws_size,
                              hipStream_t stream) {
    const int*   tokens = (const int*)  d_in[0];
    const float* emb    = (const float*)d_in[1];
    const float* wih    = (const float*)d_in[2];
    const float* whh    = (const float*)d_in[3];
    const float* bih    = (const float*)d_in[4];
    const float* bhh    = (const float*)d_in[5];
    float* out = (float*)d_out;

    char* ws = (char*)d_ws;
    _Float16* wih16 = (_Float16*)(ws + WS_WIH16);
    _Float16* whh16 = (_Float16*)(ws + WS_WHH16);
    float*    bias  = (float*)   (ws + WS_BIAS);
    _Float16* xg    = (_Float16*)(ws + WS_XG);

    prep_kernel<<<(131072 + 512 + 255) / 256, 256, 0, stream>>>(
        wih, whh, bih, bhh, wih16, whh16, bias);

    xgates_kernel<<<TT, 1024, 16384 + 2048, stream>>>(tokens, emb, wih16, bias, xg);

    scan_kernel<<<1, 1024, SMEM2, stream>>>(whh16, xg, out);
}
